// MoE_75496935129438
// MI455X (gfx1250) — matrix operand order repaired and checked by reading, not compiled, not run on hardware
//
#include <hip/hip_runtime.h>
#include <cstdint>
#include <cstddef>

// ----- problem constants (from reference) -----
static constexpr int Dm = 1024;   // model dim
static constexpr int Im = 2048;   // expert hidden (I); up-proj is 2I split even/odd
static constexpr int Em = 8;      // experts
static constexpr int Tm = 4096;   // tokens B*S

#define ALPHA_C 1.702f
#define LIMIT_C 7.0f

typedef __attribute__((ext_vector_type(16))) __bf16 v16bf;
typedef __attribute__((ext_vector_type(8)))  float  v8f;
typedef __attribute__((ext_vector_type(4)))  unsigned int u32x4;
typedef __attribute__((ext_vector_type(8)))  int i32x8;
typedef __attribute__((ext_vector_type(4)))  int i32x4;

union Frag {
    v16bf v;
    uint4 q[2];
};

#define WMMA_BF16(ACC, A, B) \
    (ACC) = __builtin_amdgcn_wmma_f32_16x16x32_bf16(false, (A), false, (B), (short)0, (ACC), false, false)

#if __has_builtin(__builtin_amdgcn_tensor_load_to_lds)
#define HAVE_TDM 1
#else
#define HAVE_TDM 0
#endif

#if __has_builtin(__builtin_amdgcn_s_wait_tensorcnt)
#define WAIT_TENSORCNT(n) __builtin_amdgcn_s_wait_tensorcnt((short)(n))
#else
#define WAIT_TENSORCNT(n)
#endif

#if HAVE_TDM
// Issue one 2D TDM tile load: tileC rows x tileK bf16 columns, row stride
// strideK (elements), from global byte address gaddr into LDS byte offset
// lds_off.  D# built per CDNA5 ISA section 8 (group0: count=1, type=2;
// group1: data_size=1 (2 bytes), tensor dims == tile dims, stride in elems).
__device__ inline void tdm_load_tile_2d(unsigned lds_off, unsigned long long gaddr,
                                        int tileK, int tileC, int strideK)
{
    u32x4 g0;
    g0[0] = 1u;                                                 // count=1 valid descriptor
    g0[1] = lds_off;                                            // lds_addr (bytes)
    g0[2] = (unsigned)(gaddr & 0xFFFFFFFFull);                  // global_addr[31:0]
    g0[3] = (unsigned)((gaddr >> 32) & 0x01FFFFFFull) | (2u << 30);  // addr[56:32] | type=2
    i32x8 g1;
    g1[0] = (int)(1u << 16);                                    // data_size=1 -> 2 bytes
    g1[1] = (int)(((unsigned)tileK & 0xFFFFu) << 16);           // tensor_dim0[15:0]
    g1[2] = (int)(((unsigned)tileC & 0xFFFFu) << 16);           // tensor_dim1[15:0]
    g1[3] = (int)(((unsigned)tileK & 0xFFFFu) << 16);           // tile_dim0
    g1[4] = (int)((unsigned)tileC & 0xFFFFu);                   // tile_dim1 (tile_dim2=0)
    g1[5] = strideK;                                            // tensor_dim0_stride[31:0]
    g1[6] = 0;                                                  // stride hi / dim1_stride lo
    g1[7] = 0;
    i32x4 z4 = {0, 0, 0, 0};
#if __clang_major__ >= 23
    i32x8 z8 = {0, 0, 0, 0, 0, 0, 0, 0};
    __builtin_amdgcn_tensor_load_to_lds(g0, g1, z4, z4, z8, 0);
#else
    __builtin_amdgcn_tensor_load_to_lds(g0, g1, z4, z4, 0);
#endif
}
#endif

// ---------------------------------------------------------------------------
// Gate: scores = softmax(x @ gate_w^T); top-2 on (scores + gate_b); dense
// combine weights comb[t][e].  One wave per token.
// ---------------------------------------------------------------------------
__global__ __launch_bounds__(256) void moe_gate_kernel(
    const float* __restrict__ x, const float* __restrict__ gw,
    const float* __restrict__ gb, float* __restrict__ comb)
{
    const int lane = threadIdx.x & 31;
    const int wid  = threadIdx.x >> 5;
    const int t    = blockIdx.x * 8 + wid;

    float acc[Em];
#pragma unroll
    for (int e = 0; e < Em; ++e) acc[e] = 0.0f;

    const float* xr = x + (size_t)t * Dm;
    for (int k = lane; k < Dm; k += 32) {
        const float xv = xr[k];
#pragma unroll
        for (int e = 0; e < Em; ++e) acc[e] += xv * gw[e * Dm + k];
    }
#pragma unroll
    for (int e = 0; e < Em; ++e) {
#pragma unroll
        for (int off = 16; off > 0; off >>= 1)
            acc[e] += __shfl_xor(acc[e], off, 32);
    }
    if (lane == 0) {
        float m = acc[0];
#pragma unroll
        for (int e = 1; e < Em; ++e) m = fmaxf(m, acc[e]);
        float s = 0.0f, sc[Em];
#pragma unroll
        for (int e = 0; e < Em; ++e) { sc[e] = __expf(acc[e] - m); s += sc[e]; }
        const float inv = 1.0f / s;
        float biased[Em];
#pragma unroll
        for (int e = 0; e < Em; ++e) { sc[e] *= inv; biased[e] = sc[e] + gb[e]; }
        int i1 = 0;
#pragma unroll
        for (int e = 1; e < Em; ++e) if (biased[e] > biased[i1]) i1 = e;
        int i2 = (i1 == 0) ? 1 : 0;
#pragma unroll
        for (int e = 0; e < Em; ++e)
            if (e != i1 && e != i2 && biased[e] > biased[i2]) i2 = e;
#pragma unroll
        for (int e = 0; e < Em; ++e)
            comb[(size_t)t * Em + e] = (e == i1) ? sc[i1] : ((e == i2) ? sc[i2] : 0.0f);
    }
}

// ---------------------------------------------------------------------------
__global__ __launch_bounds__(256) void cast_bf16_kernel(
    const float* __restrict__ in, __bf16* __restrict__ out, int n)
{
    int i = (blockIdx.x * 256 + threadIdx.x) * 4;
    if (i + 3 < n) {
        const float4 v = *(const float4*)(in + i);
        out[i + 0] = (__bf16)v.x;
        out[i + 1] = (__bf16)v.y;
        out[i + 2] = (__bf16)v.z;
        out[i + 3] = (__bf16)v.w;
    }
}

__global__ __launch_bounds__(256) void zero_f32_kernel(float* __restrict__ p, int n)
{
    int i = (blockIdx.x * 256 + threadIdx.x) * 4;
    if (i + 3 < n) *(float4*)(p + i) = make_float4(0.f, 0.f, 0.f, 0.f);
}

// ---------------------------------------------------------------------------
// Transpose+convert, splitting even/odd columns (w1/w3):
//   in : K x (2C) fp32 ; oute[c][k]=in[k][2c] ; outo[c][k]=in[k][2c+1]  (bf16)
// ---------------------------------------------------------------------------
__global__ __launch_bounds__(256) void transpose_split_kernel(
    const float* __restrict__ in, __bf16* __restrict__ oute,
    __bf16* __restrict__ outo, int K, int C)
{
    __shared__ float te[32][33];
    __shared__ float to[32][33];
    const int k0 = blockIdx.x * 32, c0 = blockIdx.y * 32;
    const int tx = threadIdx.x & 31, ty = threadIdx.x >> 5;
#pragma unroll
    for (int r = ty; r < 32; r += 8) {
        const float2 p = *(const float2*)(in + (size_t)(k0 + r) * (2 * C) + 2 * (c0 + tx));
        te[r][tx] = p.x;
        to[r][tx] = p.y;
    }
    __syncthreads();
#pragma unroll
    for (int r = ty; r < 32; r += 8) {
        const size_t oidx = (size_t)(c0 + r) * K + k0 + tx;
        oute[oidx] = (__bf16)te[tx][r];
        outo[oidx] = (__bf16)to[tx][r];
    }
}

// Plain transpose+convert: in R x C fp32 -> out C x R bf16  (w2 style)
__global__ __launch_bounds__(256) void transpose_kernel(
    const float* __restrict__ in, __bf16* __restrict__ out, int R, int C)
{
    __shared__ float t[32][33];
    const int r0 = blockIdx.x * 32, c0 = blockIdx.y * 32;
    const int tx = threadIdx.x & 31, ty = threadIdx.x >> 5;
#pragma unroll
    for (int r = ty; r < 32; r += 8)
        t[r][tx] = in[(size_t)(r0 + r) * C + c0 + tx];
    __syncthreads();
#pragma unroll
    for (int r = ty; r < 32; r += 8)
        out[(size_t)(c0 + r) * R + r0 + tx] = (__bf16)t[tx][r];
}

// ---------------------------------------------------------------------------
// swiglu epilogue helper (even channel = gate, odd channel = linear)
// ---------------------------------------------------------------------------
__device__ inline __bf16 swiglu_elem(float v1e, float v1o, float v3e, float v3o)
{
    const float g = fminf(v1e * v3e, LIMIT_C);
    const float l = fminf(fmaxf(v1o * v3o, -LIMIT_C), LIMIT_C);
    const float s = 1.0f / (1.0f + __expf(-ALPHA_C * g));
    return (__bf16)(g * s * (l + 1.0f));
}

#if HAVE_TDM
// ---------------------------------------------------------------------------
// GEMM1 (TDM version): H = swiglu((X@w1+b1)*(X@w3+b3)).
// B tiles (4 matrices x 32 cols x 64 K bf16 = 16 KB) streamed Global->LDS by
// the Tensor Data Mover, double-buffered; wave 0 issues DMAs and gates with
// s_wait_tensorcnt; all waves read B fragments from LDS.  A stays on the
// direct global path (L2-resident).  8 v_wmma per 32-deep K step.
// ---------------------------------------------------------------------------
__global__ __launch_bounds__(256) void moe_gemm1_tdm_kernel(
    const __bf16* __restrict__ X,
    const __bf16* __restrict__ W1e, const __bf16* __restrict__ W1o,
    const __bf16* __restrict__ W3e, const __bf16* __restrict__ W3o,
    const float* __restrict__ B1, const float* __restrict__ B3,
    __bf16* __restrict__ H)
{
    constexpr int KC = 64;                       // K-chunk per TDM tile
    constexpr int TILE_ELEMS = 32 * KC;          // 32 cols x 64 K
    __shared__ __align__(32) __bf16 bt[2][4][TILE_ELEMS];

    const int lane   = threadIdx.x & 31;
    const int wid    = threadIdx.x >> 5;
    const int laneHi = lane >> 4;
    const int lanelo = lane & 15;

    const int m0   = blockIdx.y * 128 + (wid & 3) * 32;
    const int cblk = blockIdx.x * 32;
    const int cw   = (wid >> 2) * 16;

    v8f a1e0 = {}, a1e1 = {}, a1o0 = {}, a1o1 = {};
    v8f a3e0 = {}, a3e1 = {}, a3o0 = {}, a3o1 = {};

    const __bf16* xp0 = X + (size_t)(m0 + lanelo) * Dm + laneHi * 16;
    const __bf16* xp1 = X + (size_t)(m0 + 16 + lanelo) * Dm + laneHi * 16;

    const unsigned long long ga[4] = {
        (unsigned long long)(uintptr_t)(W1e + (size_t)cblk * Dm),
        (unsigned long long)(uintptr_t)(W1o + (size_t)cblk * Dm),
        (unsigned long long)(uintptr_t)(W3e + (size_t)cblk * Dm),
        (unsigned long long)(uintptr_t)(W3o + (size_t)cblk * Dm)};
    const unsigned ldsb = (unsigned)(size_t)(&bt[0][0][0]);

    // prologue: stream chunk 0 into buffer 0
    if (wid == 0) {
#pragma unroll
        for (int m = 0; m < 4; ++m)
            tdm_load_tile_2d(ldsb + (unsigned)(m * TILE_ELEMS * 2), ga[m], KC, 32, Dm);
    }

    constexpr int NCHUNK = Dm / KC;
    for (int n = 0; n < NCHUNK; ++n) {
        const int buf = n & 1;
        if (wid == 0) {
            if (n + 1 < NCHUNK) {
                const unsigned lo = ldsb + (unsigned)((((n + 1) & 1) * 4) * TILE_ELEMS * 2);
#pragma unroll
                for (int m = 0; m < 4; ++m)
                    tdm_load_tile_2d(lo + (unsigned)(m * TILE_ELEMS * 2),
                                     ga[m] + (unsigned long long)(n + 1) * KC * 2, KC, 32, Dm);
                WAIT_TENSORCNT(4);   // chunk n complete (in-order), n+1 in flight
            } else {
                WAIT_TENSORCNT(0);
            }
        }
        __syncthreads();             // chunk n visible to all waves

        const int kg = n * KC;
        const int brow = cw + lanelo;
#pragma unroll
        for (int ks = 0; ks < 2; ++ks) {
            Frag a0, a1, b1e, b1o, b3e, b3o;
            a0.q[0] = *(const uint4*)(xp0 + kg + ks * 32);
            a0.q[1] = *(const uint4*)(xp0 + kg + ks * 32 + 8);
            a1.q[0] = *(const uint4*)(xp1 + kg + ks * 32);
            a1.q[1] = *(const uint4*)(xp1 + kg + ks * 32 + 8);
            const int bofs = brow * KC + ks * 32 + laneHi * 16;
            b1e.v = *(const v16bf*)(&bt[buf][0][bofs]);
            b1o.v = *(const v16bf*)(&bt[buf][1][bofs]);
            b3e.v = *(const v16bf*)(&bt[buf][2][bofs]);
            b3o.v = *(const v16bf*)(&bt[buf][3][bofs]);
            WMMA_BF16(a1e0, a0.v, b1e.v);
            WMMA_BF16(a1e1, a1.v, b1e.v);
            WMMA_BF16(a1o0, a0.v, b1o.v);
            WMMA_BF16(a1o1, a1.v, b1o.v);
            WMMA_BF16(a3e0, a0.v, b3e.v);
            WMMA_BF16(a3e1, a1.v, b3e.v);
            WMMA_BF16(a3o0, a0.v, b3o.v);
            WMMA_BF16(a3o1, a1.v, b3o.v);
        }
        __syncthreads();             // all reads done before buf is re-filled
    }

    const int c = cblk + cw + lanelo;
    const float b1ev = B1[2 * c], b1ov = B1[2 * c + 1];
    const float b3ev = B3[2 * c], b3ov = B3[2 * c + 1];
    const int mofs = laneHi * 8;
#pragma unroll
    for (int i = 0; i < 8; ++i) {
        H[(size_t)(m0 + mofs + i) * Im + c] =
            swiglu_elem(a1e0[i] + b1ev, a1o0[i] + b1ov, a3e0[i] + b3ev, a3o0[i] + b3ov);
        H[(size_t)(m0 + 16 + mofs + i) * Im + c] =
            swiglu_elem(a1e1[i] + b1ev, a1o1[i] + b1ov, a3e1[i] + b3ev, a3o1[i] + b3ov);
    }
}
#endif  // HAVE_TDM

// ---------------------------------------------------------------------------
// GEMM1 (direct-global fallback) — proven codegen from round 1.
// ---------------------------------------------------------------------------
__global__ __launch_bounds__(256) void moe_gemm1_kernel(
    const __bf16* __restrict__ X,
    const __bf16* __restrict__ W1e, const __bf16* __restrict__ W1o,
    const __bf16* __restrict__ W3e, const __bf16* __restrict__ W3o,
    const float* __restrict__ B1, const float* __restrict__ B3,
    __bf16* __restrict__ H)
{
    const int lane   = threadIdx.x & 31;
    const int wid    = threadIdx.x >> 5;
    const int laneHi = lane >> 4;
    const int lanelo = lane & 15;

    const int m0 = blockIdx.y * 128 + (wid & 3) * 32;
    const int c0 = blockIdx.x * 32  + (wid >> 2) * 16;

    v8f a1e0 = {}, a1e1 = {}, a1o0 = {}, a1o1 = {};
    v8f a3e0 = {}, a3e1 = {}, a3o0 = {}, a3o1 = {};

    const __bf16* xp0 = X + (size_t)(m0 + lanelo) * Dm + laneHi * 16;
    const __bf16* xp1 = X + (size_t)(m0 + 16 + lanelo) * Dm + laneHi * 16;
    const size_t wro = (size_t)(c0 + lanelo) * Dm + laneHi * 16;
    const __bf16* w1ep = W1e + wro;
    const __bf16* w1op = W1o + wro;
    const __bf16* w3ep = W3e + wro;
    const __bf16* w3op = W3o + wro;

    for (int k0 = 0; k0 < Dm; k0 += 32) {
        __builtin_prefetch(w1ep + k0 + 128, 0, 1);
        __builtin_prefetch(w3ep + k0 + 128, 0, 1);
        Frag a0, a1, b1e, b1o, b3e, b3o;
        a0.q[0] = *(const uint4*)(xp0 + k0);
        a0.q[1] = *(const uint4*)(xp0 + k0 + 8);
        a1.q[0] = *(const uint4*)(xp1 + k0);
        a1.q[1] = *(const uint4*)(xp1 + k0 + 8);
        b1e.v = *(const v16bf*)(w1ep + k0);
        b1o.v = *(const v16bf*)(w1op + k0);
        b3e.v = *(const v16bf*)(w3ep + k0);
        b3o.v = *(const v16bf*)(w3op + k0);
        WMMA_BF16(a1e0, a0.v, b1e.v);
        WMMA_BF16(a1e1, a1.v, b1e.v);
        WMMA_BF16(a1o0, a0.v, b1o.v);
        WMMA_BF16(a1o1, a1.v, b1o.v);
        WMMA_BF16(a3e0, a0.v, b3e.v);
        WMMA_BF16(a3e1, a1.v, b3e.v);
        WMMA_BF16(a3o0, a0.v, b3o.v);
        WMMA_BF16(a3o1, a1.v, b3o.v);
    }

    const int c = c0 + lanelo;
    const float b1ev = B1[2 * c], b1ov = B1[2 * c + 1];
    const float b3ev = B3[2 * c], b3ov = B3[2 * c + 1];
    const int mofs = laneHi * 8;
#pragma unroll
    for (int i = 0; i < 8; ++i) {
        H[(size_t)(m0 + mofs + i) * Im + c] =
            swiglu_elem(a1e0[i] + b1ev, a1o0[i] + b1ov, a3e0[i] + b3ev, a3o0[i] + b3ov);
        H[(size_t)(m0 + 16 + mofs + i) * Im + c] =
            swiglu_elem(a1e1[i] + b1ev, a1o1[i] + b1ov, a3e1[i] + b3ev, a3o1[i] + b3ov);
    }
}

// ---------------------------------------------------------------------------
// GEMM2: OUT[t][d] += comb[t][e] * (H @ w2 + b2)   (comb==nullptr -> 1.0)
// Wave tile 32x32, block 64x128.  4 v_wmma per 32-deep K step.
// ---------------------------------------------------------------------------
__global__ __launch_bounds__(256) void moe_gemm2_kernel(
    const __bf16* __restrict__ Hm, const __bf16* __restrict__ W2T,
    const float* __restrict__ B2, const float* __restrict__ comb,
    int expert, float* __restrict__ OUT)
{
    const int lane   = threadIdx.x & 31;
    const int wid    = threadIdx.x >> 5;
    const int laneHi = lane >> 4;
    const int lanelo = lane & 15;

    const int m0 = blockIdx.y * 64  + (wid & 1) * 32;
    const int c0 = blockIdx.x * 128 + (wid >> 1) * 32;

    v8f acc00 = {}, acc01 = {}, acc10 = {}, acc11 = {};

    const __bf16* hp0 = Hm + (size_t)(m0 + lanelo) * Im + laneHi * 16;
    const __bf16* hp1 = Hm + (size_t)(m0 + 16 + lanelo) * Im + laneHi * 16;
    const __bf16* wp0 = W2T + (size_t)(c0 + lanelo) * Im + laneHi * 16;
    const __bf16* wp1 = W2T + (size_t)(c0 + 16 + lanelo) * Im + laneHi * 16;

    for (int k0 = 0; k0 < Im; k0 += 32) {
        __builtin_prefetch(wp0 + k0 + 128, 0, 1);
        Frag a0, a1, b0, b1;
        a0.q[0] = *(const uint4*)(hp0 + k0);
        a0.q[1] = *(const uint4*)(hp0 + k0 + 8);
        a1.q[0] = *(const uint4*)(hp1 + k0);
        a1.q[1] = *(const uint4*)(hp1 + k0 + 8);
        b0.v = *(const v16bf*)(wp0 + k0);
        b1.v = *(const v16bf*)(wp1 + k0);
        WMMA_BF16(acc00, a0.v, b0.v);
        WMMA_BF16(acc01, a0.v, b1.v);
        WMMA_BF16(acc10, a1.v, b0.v);
        WMMA_BF16(acc11, a1.v, b1.v);
    }

    const int d0 = c0 + lanelo, d1 = c0 + 16 + lanelo;
    const float bb0 = B2[d0], bb1 = B2[d1];
    const int mofs = laneHi * 8;
#pragma unroll
    for (int i = 0; i < 8; ++i) {
        const int ma = m0 + mofs + i;
        const int mb = ma + 16;
        const float ca = comb ? comb[(size_t)ma * Em + expert] : 1.0f;
        const float cb = comb ? comb[(size_t)mb * Em + expert] : 1.0f;
        OUT[(size_t)ma * Dm + d0] += ca * (acc00[i] + bb0);
        OUT[(size_t)ma * Dm + d1] += ca * (acc01[i] + bb1);
        OUT[(size_t)mb * Dm + d0] += cb * (acc10[i] + bb0);
        OUT[(size_t)mb * Dm + d1] += cb * (acc11[i] + bb1);
    }
}

// ---------------------------------------------------------------------------
extern "C" void kernel_launch(void* const* d_in, const int* in_sizes, int n_in,
                              void* d_out, int out_size, void* d_ws, size_t ws_size,
                              hipStream_t stream)
{
    (void)in_sizes; (void)n_in; (void)out_size; (void)ws_size;
    const float* x      = (const float*)d_in[0];
    const float* gate_w = (const float*)d_in[1];
    const float* gate_b = (const float*)d_in[2];
    const float* w1     = (const float*)d_in[3];
    const float* b1     = (const float*)d_in[4];
    const float* w3     = (const float*)d_in[5];
    const float* b3     = (const float*)d_in[6];
    const float* w2     = (const float*)d_in[7];
    const float* b2     = (const float*)d_in[8];
    const float* sw1    = (const float*)d_in[9];
    const float* sb1    = (const float*)d_in[10];
    const float* sw3    = (const float*)d_in[11];
    const float* sb3    = (const float*)d_in[12];
    const float* sw2    = (const float*)d_in[13];
    const float* sb2    = (const float*)d_in[14];
    float* out = (float*)d_out;

    char* ws = (char*)d_ws;
    size_t off = 0;
    auto alloc = [&](size_t bytes) -> void* {
        void* p = (void*)(ws + off);
        off += (bytes + 255) & ~((size_t)255);
        return p;
    };

    float*  comb = (float*) alloc((size_t)Tm * Em * 4);
    __bf16* xb   = (__bf16*)alloc((size_t)Tm * Dm * 2);
    __bf16* hb   = (__bf16*)alloc((size_t)Tm * Im * 2);
    __bf16* w1eT = (__bf16*)alloc((size_t)Em * Im * Dm * 2);
    __bf16* w1oT = (__bf16*)alloc((size_t)Em * Im * Dm * 2);
    __bf16* w3eT = (__bf16*)alloc((size_t)Em * Im * Dm * 2);
    __bf16* w3oT = (__bf16*)alloc((size_t)Em * Im * Dm * 2);
    __bf16* w2T  = (__bf16*)alloc((size_t)Em * Dm * Im * 2);
    __bf16* s1eT = (__bf16*)alloc((size_t)Im * Dm * 2);
    __bf16* s1oT = (__bf16*)alloc((size_t)Im * Dm * 2);
    __bf16* s3eT = (__bf16*)alloc((size_t)Im * Dm * 2);
    __bf16* s3oT = (__bf16*)alloc((size_t)Im * Dm * 2);
    __bf16* s2T  = (__bf16*)alloc((size_t)Dm * Im * 2);

    // 1) gating -> dense combine weights
    moe_gate_kernel<<<Tm / 8, 256, 0, stream>>>(x, gate_w, gate_b, comb);

    // 2) cast activations to bf16
    cast_bf16_kernel<<<(Tm * Dm / 4 + 255) / 256, 256, 0, stream>>>(x, xb, Tm * Dm);

    // 3) convert+transpose weights into WMMA-friendly N-major bf16
    const dim3 tsGrid(Dm / 32, Im / 32);
    const dim3 t2Grid(Im / 32, Dm / 32);
    for (int e = 0; e < Em; ++e) {
        const size_t wofs = (size_t)e * Dm * 2 * Im;
        const size_t woT  = (size_t)e * Im * Dm;
        transpose_split_kernel<<<tsGrid, 256, 0, stream>>>(w1 + wofs, w1eT + woT, w1oT + woT, Dm, Im);
        transpose_split_kernel<<<tsGrid, 256, 0, stream>>>(w3 + wofs, w3eT + woT, w3oT + woT, Dm, Im);
        transpose_kernel<<<t2Grid, 256, 0, stream>>>(w2 + (size_t)e * Im * Dm, w2T + (size_t)e * Dm * Im, Im, Dm);
    }
    transpose_split_kernel<<<tsGrid, 256, 0, stream>>>(sw1, s1eT, s1oT, Dm, Im);
    transpose_split_kernel<<<tsGrid, 256, 0, stream>>>(sw3, s3eT, s3oT, Dm, Im);
    transpose_kernel<<<t2Grid, 256, 0, stream>>>(sw2, s2T, Im, Dm);

    // 4) zero output accumulator
    zero_f32_kernel<<<(Tm * Dm / 4 + 255) / 256, 256, 0, stream>>>(out, Tm * Dm);

    // 5) experts: GEMM1 (fused swiglu) then GEMM2 (scaled accumulate)
    const dim3 g1Grid(Im / 32, Tm / 128);
    const dim3 g2Grid(Dm / 128, Tm / 64);
    for (int e = 0; e < Em; ++e) {
        const size_t woT = (size_t)e * Im * Dm;
#if HAVE_TDM
        moe_gemm1_tdm_kernel<<<g1Grid, 256, 0, stream>>>(
            xb, w1eT + woT, w1oT + woT, w3eT + woT, w3oT + woT,
            b1 + (size_t)e * 2 * Im, b3 + (size_t)e * 2 * Im, hb);
#else
        moe_gemm1_kernel<<<g1Grid, 256, 0, stream>>>(
            xb, w1eT + woT, w1oT + woT, w3eT + woT, w3oT + woT,
            b1 + (size_t)e * 2 * Im, b3 + (size_t)e * 2 * Im, hb);
#endif
        moe_gemm2_kernel<<<g2Grid, 256, 0, stream>>>(
            hb, w2T + (size_t)e * Dm * Im, b2 + (size_t)e * Dm, comb, e, out);
    }

    // 6) shared expert (combine weight 1.0)
#if HAVE_TDM
    moe_gemm1_tdm_kernel<<<g1Grid, 256, 0, stream>>>(xb, s1eT, s1oT, s3eT, s3oT, sb1, sb3, hb);
#else
    moe_gemm1_kernel<<<g1Grid, 256, 0, stream>>>(xb, s1eT, s1oT, s3eT, s3oT, sb1, sb3, hb);
#endif
    moe_gemm2_kernel<<<g2Grid, 256, 0, stream>>>(hb, s2T, sb2, nullptr, 0, out);
}
